// MLPDecoder_10196252360963
// MI455X (gfx1250) — matrix operand order repaired and checked by reading, not compiled, not run on hardware
//
#include <hip/hip_runtime.h>
#include <hip/hip_bf16.h>

#define N_NODES 2048
#define N_EDGES 32768
#define N_IN    256
#define KDIM    512   // every GEMM in this net has K = 512

typedef __attribute__((ext_vector_type(16))) _Float16 v16h;
typedef __attribute__((ext_vector_type(8)))  _Float16 v8h;
typedef __attribute__((ext_vector_type(8)))  float    v8f;

// ---------------------------------------------------------------------------
// Kernel 1: recover edge indices from the one-hot incidence matrices.
// One block per row (2E blocks). 256 threads x 8 floats = 2048 cols, coalesced.
// ---------------------------------------------------------------------------
__global__ __launch_bounds__(256) void k_extract_idx(
    const float* __restrict__ rel_rec, const float* __restrict__ rel_send,
    int* __restrict__ recv_idx, int* __restrict__ send_idx)
{
    __shared__ int s_idx;
    int row = blockIdx.x;
    const float* base;
    int* out;
    if (row < N_EDGES) { base = rel_rec + (size_t)row * N_NODES;              out = recv_idx + row; }
    else               { base = rel_send + (size_t)(row - N_EDGES) * N_NODES; out = send_idx + (row - N_EDGES); }
    if (threadIdx.x == 0) s_idx = 0;
    __syncthreads();
    int t = threadIdx.x;
    int found = -1;
#pragma unroll
    for (int i = 0; i < 2; ++i) {
        int col4 = t + 256 * i;                 // float4 index
        float4 v = *(const float4*)(base + col4 * 4);
        if (v.x > 0.5f) found = col4 * 4 + 0;
        if (v.y > 0.5f) found = col4 * 4 + 1;
        if (v.z > 0.5f) found = col4 * 4 + 2;
        if (v.w > 0.5f) found = col4 * 4 + 3;
    }
    if (found >= 0) atomicMax(&s_idx, found);
    __syncthreads();
    if (t == 0) *out = s_idx;
}

// ---------------------------------------------------------------------------
// Kernel 2: fp32 -> fp16 convert (inputs + all weight matrices)
// ---------------------------------------------------------------------------
__global__ void k_cvt_f16(const float* __restrict__ src, _Float16* __restrict__ dst, int n)
{
    int i = blockIdx.x * blockDim.x + threadIdx.x;
    if (i < n) dst[i] = (_Float16)src[i];
}

__global__ void k_zero_f32(float* __restrict__ p, int n)
{
    int i = blockIdx.x * blockDim.x + threadIdx.x;
    if (i < n) p[i] = 0.0f;
}

// ---------------------------------------------------------------------------
// Kernel 3: gather: pre_msg[e] = [ xh[send[e]], xh[recv[e]] ]  (f16, uint copies)
// ---------------------------------------------------------------------------
__global__ __launch_bounds__(128) void k_build_pre_msg(
    const _Float16* __restrict__ xh, const int* __restrict__ recv_idx,
    const int* __restrict__ send_idx, _Float16* __restrict__ pre_msg)
{
    int e = blockIdx.x, t = threadIdx.x;           // 128 threads; row = 256 f16 = 128 u32
    const unsigned* xs = (const unsigned*)(xh + (size_t)send_idx[e] * N_IN);
    const unsigned* xr = (const unsigned*)(xh + (size_t)recv_idx[e] * N_IN);
    unsigned* o = (unsigned*)(pre_msg + (size_t)e * (2 * N_IN));
    o[t]       = xs[t];
    o[128 + t] = xr[t];
}

// ---------------------------------------------------------------------------
// Kernel 4: aug[n] = [ xh[n], f16(agg[n]) ]
// ---------------------------------------------------------------------------
__global__ __launch_bounds__(128) void k_build_aug(
    const _Float16* __restrict__ xh, const float* __restrict__ agg,
    _Float16* __restrict__ aug)
{
    int n = blockIdx.x, t = threadIdx.x;           // 128 threads
    unsigned* o = (unsigned*)(aug + (size_t)n * 512);
    o[t] = ((const unsigned*)(xh + (size_t)n * N_IN))[t];
    float2 g = *(const float2*)(agg + (size_t)n * N_IN + 2 * t);
    union { _Float16 h[2]; unsigned u; } p;
    p.h[0] = (_Float16)g.x;
    p.h[1] = (_Float16)g.y;
    o[128 + t] = p.u;
}

// ---------------------------------------------------------------------------
// WMMA GEMM:  C[M,N] = epilogue( A[M,K=512] @ W[N,K]^T + bias )
// One wave computes a 32x64 tile: 8 x v8f accumulators; each 32B B fragment
// feeds two WMMAs (24 B loaded per wmma). K-loop: 16 steps x 8 wmma.
// EP==0: ReLU, store f16    EP==1: ReLU, scatter atomicAdd f32 by idx[m]
// EP==2: + residual, store f32
// ---------------------------------------------------------------------------
template <int EP>
__global__ __launch_bounds__(128) void k_gemm512(
    const _Float16* __restrict__ A, const _Float16* __restrict__ W,
    const float* __restrict__ bias, int M, int N,
    _Float16* __restrict__ outh, float* __restrict__ outf,
    const int* __restrict__ idx, const float* __restrict__ resid)
{
    const int wave = threadIdx.x >> 5;
    const int lane = threadIdx.x & 31;
    const int tiles_m = M >> 5;                    // 32-row tiles
    const int tiles_n = N >> 6;                    // 64-col tiles
    const int tile = blockIdx.x * 4 + wave;
    if (tile >= tiles_m * tiles_n) return;         // wave-uniform
    const int mt = tile % tiles_m;                 // m-fastest: 4 waves of a block
    const int nt = tile / tiles_m;                 // share the same W columns
    const int m0 = mt << 5, n0 = nt << 6;

    const int mi = lane & 15;                      // A row / B column / C column within tile
    const int q  = lane >> 4;                      // half-wave selector

    // A: row-major f16 [M,512]; lane holds K = 8q..8q+7 and 8q+16..8q+23 per 32-chunk
    const _Float16* arow0 = A + (size_t)(m0 + mi) * KDIM + 8 * q;
    const _Float16* arow1 = arow0 + (size_t)16 * KDIM;      // rows m0+16..m0+31
    // B: W row-major [N,512]; lane holds column (n0+j*16+mi), K = 8q..8q+7 and 8q+16..8q+23 per chunk
    const _Float16* wbase = W + (size_t)(n0 + mi) * KDIM + 8 * q;

    v8f c[8];
#pragma unroll
    for (int j = 0; j < 8; ++j) c[j] = (v8f){0.f, 0.f, 0.f, 0.f, 0.f, 0.f, 0.f, 0.f};

    for (int k0 = 0; k0 < KDIM; k0 += 32) {
        union { v16h v; v8h h[2]; } a0, a1;
        a0.h[0] = *(const v8h*)(arow0 + k0);
        a0.h[1] = *(const v8h*)(arow0 + k0 + 16);
        a1.h[0] = *(const v8h*)(arow1 + k0);
        a1.h[1] = *(const v8h*)(arow1 + k0 + 16);
#pragma unroll
        for (int j = 0; j < 4; ++j) {
            union { v16h v; v8h h[2]; } bu;
            bu.h[0] = *(const v8h*)(wbase + (size_t)(j * 16) * KDIM + k0);
            bu.h[1] = *(const v8h*)(wbase + (size_t)(j * 16) * KDIM + k0 + 16);
            v16h b = bu.v;
            c[j]     = __builtin_amdgcn_wmma_f32_16x16x32_f16(
                false, a0.v, false, b, (short)0, c[j], false, false);
            c[4 + j] = __builtin_amdgcn_wmma_f32_16x16x32_f16(
                false, a1.v, false, b, (short)0, c[4 + j], false, false);
        }
    }

#pragma unroll
    for (int j = 0; j < 4; ++j) {
        const int n = n0 + j * 16 + mi;            // C: column = lane&15
        const float bv = bias[n];
#pragma unroll
        for (int half = 0; half < 2; ++half) {
            const v8f cf = c[4 * half + j];
#pragma unroll
            for (int r = 0; r < 8; ++r) {
                const int m = m0 + 16 * half + r + 8 * q;   // C row = vgpr + 8*(lane>>4)
                float v = cf[r] + bv;
                if (EP != 2) v = v > 0.f ? v : 0.f;
                if (EP == 0) {
                    outh[(size_t)m * N + n] = (_Float16)v;
                } else if (EP == 1) {
                    atomicAdd(&outf[(size_t)idx[m] * N + n], v);
                } else {
                    outf[(size_t)m * N + n] = v + resid[(size_t)m * N + n];
                }
            }
        }
    }
}

// ---------------------------------------------------------------------------
// Host-side orchestration
// ---------------------------------------------------------------------------
extern "C" void kernel_launch(void* const* d_in, const int* in_sizes, int n_in,
                              void* d_out, int out_size, void* d_ws, size_t ws_size,
                              hipStream_t stream)
{
    const float* inputs = (const float*)d_in[0];   // [2048,256]
    const float* rel_rec  = (const float*)d_in[1]; // [E,N]
    const float* rel_send = (const float*)d_in[2]; // [E,N]
    const float* msg_W1 = (const float*)d_in[3];   const float* msg_b1 = (const float*)d_in[4];
    const float* msg_W2 = (const float*)d_in[5];   const float* msg_b2 = (const float*)d_in[6];
    const float* out_W1 = (const float*)d_in[7];   const float* out_b1 = (const float*)d_in[8];
    const float* out_W2 = (const float*)d_in[9];   const float* out_b2 = (const float*)d_in[10];
    const float* out_W3 = (const float*)d_in[11];  const float* out_b3 = (const float*)d_in[12];
    float* out = (float*)d_out;

    char* ws = (char*)d_ws;
    size_t off = 0;
    auto alloc = [&](size_t bytes) { char* p = ws + off; off += (bytes + 255) & ~(size_t)255; return p; };

    int*      recv_idx = (int*)alloc(N_EDGES * sizeof(int));
    int*      send_idx = (int*)alloc(N_EDGES * sizeof(int));
    _Float16* xh   = (_Float16*)alloc((size_t)N_NODES * N_IN * 2);
    _Float16* W1h  = (_Float16*)alloc(512 * 512 * 2);
    _Float16* W2h  = (_Float16*)alloc(256 * 512 * 2);
    _Float16* OW1h = (_Float16*)alloc(512 * 512 * 2);
    _Float16* OW2h = (_Float16*)alloc(512 * 512 * 2);
    _Float16* OW3h = (_Float16*)alloc(256 * 512 * 2);
    float*    agg  = (float*)alloc((size_t)N_NODES * 256 * 4);
    _Float16* aug  = (_Float16*)alloc((size_t)N_NODES * 512 * 2);
    _Float16* hid1 = (_Float16*)alloc((size_t)N_NODES * 512 * 2);
    _Float16* hid2 = (_Float16*)alloc((size_t)N_NODES * 512 * 2);
    _Float16* pre_msg = (_Float16*)alloc((size_t)N_EDGES * 512 * 2);
    _Float16* hidden  = (_Float16*)alloc((size_t)N_EDGES * 512 * 2);

    // 1) one-hot -> indices (single 512MB streaming pass; ~22us HBM floor)
    k_extract_idx<<<2 * N_EDGES, 256, 0, stream>>>(rel_rec, rel_send, recv_idx, send_idx);

    // 2) fp32 -> fp16 conversions (inputs + weights)
    auto cvt = [&](const float* s, _Float16* d, int n) {
        k_cvt_f16<<<(n + 255) / 256, 256, 0, stream>>>(s, d, n);
    };
    cvt(inputs, xh,   N_NODES * N_IN);
    cvt(msg_W1, W1h,  512 * 512);
    cvt(msg_W2, W2h,  256 * 512);
    cvt(out_W1, OW1h, 512 * 512);
    cvt(out_W2, OW2h, 512 * 512);
    cvt(out_W3, OW3h, 256 * 512);

    // 3) zero scatter accumulator (harness does not re-zero between replays)
    k_zero_f32<<<(N_NODES * 256 + 255) / 256, 256, 0, stream>>>(agg, N_NODES * 256);

    // 4) gather node features to edges
    k_build_pre_msg<<<N_EDGES, 128, 0, stream>>>(xh, recv_idx, send_idx, pre_msg);

    // 5) edge MLP layer 1: hidden = relu(pre_msg @ W1^T + b1)   [32768 x 512]
    {
        int tiles = (N_EDGES / 32) * (512 / 64);
        k_gemm512<0><<<tiles / 4, 128, 0, stream>>>(pre_msg, W1h, msg_b1,
            N_EDGES, 512, hidden, nullptr, nullptr, nullptr);
    }
    // 6) edge MLP layer 2 + fused scatter-add: agg[recv[e]] += relu(hidden @ W2^T + b2)
    {
        int tiles = (N_EDGES / 32) * (256 / 64);
        k_gemm512<1><<<tiles / 4, 128, 0, stream>>>(hidden, W2h, msg_b2,
            N_EDGES, 256, nullptr, agg, recv_idx, nullptr);
    }
    // 7) aug = [inputs, agg]
    k_build_aug<<<N_NODES, 128, 0, stream>>>(xh, agg, aug);

    // 8) node MLP layer 1: hid1 = relu(aug @ out_W1^T + b1)     [2048 x 512]
    {
        int tiles = (N_NODES / 32) * (512 / 64);
        k_gemm512<0><<<tiles / 4, 128, 0, stream>>>(aug, OW1h, out_b1,
            N_NODES, 512, hid1, nullptr, nullptr, nullptr);
    }
    // 9) node MLP layer 2: hid2 = relu(hid1 @ out_W2^T + b2)
    {
        int tiles = (N_NODES / 32) * (512 / 64);
        k_gemm512<0><<<tiles / 4, 128, 0, stream>>>(hid1, OW2h, out_b2,
            N_NODES, 512, hid2, nullptr, nullptr, nullptr);
    }
    // 10) final: out = inputs + (hid2 @ out_W3^T + b3)          [2048 x 256] fp32
    {
        int tiles = (N_NODES / 32) * (256 / 64);
        k_gemm512<2><<<tiles / 4, 128, 0, stream>>>(hid2, OW3h, out_b3,
            N_NODES, 256, nullptr, out, nullptr, inputs);
    }
    (void)in_sizes; (void)n_in; (void)out_size; (void)ws_size;
}
